// SphereRayTracer_60404420051261
// MI455X (gfx1250) — matrix operand order repaired and checked by reading, not compiled, not run on hardware
//
#include <hip/hip_runtime.h>

#define THR     5e-05f
#define NRAYS   32768
#define HID     128

typedef __attribute__((ext_vector_type(16))) _Float16 v16h;
typedef __attribute__((ext_vector_type(8)))  _Float16 v8h;
typedef __attribute__((ext_vector_type(8)))  float    v8f;

#define LOG2E 1.4426950408889634f
#define LN2   0.6931471805599453f

// Fast numerically-stable softplus: max(x,0) + ln2*log2(1 + 2^(-|x|*log2e)).
// Log argument is in (1,2] -> raw v_exp_f32 / v_log_f32 need no denorm guards.
__device__ __forceinline__ float softplus_f(float x) {
    const float t = __builtin_amdgcn_exp2f(-fabsf(x) * LOG2E);
    return fmaf(LN2, __builtin_amdgcn_logf(1.0f + t), fmaxf(x, 0.0f));
}

// One-time: convert W2 (128x128 f32 row-major [k][n]) to f16 transposed [n][k]
__global__ void __launch_bounds__(256)
prep_w2t_kernel(const float* __restrict__ W2, _Float16* __restrict__ w2t) {
    int idx = blockIdx.x * 256 + threadIdx.x;   // 0..16383
    int n = idx >> 7;
    int k = idx & 127;
    w2t[idx] = (_Float16)W2[k * HID + n];
}

__device__ __forceinline__ v8f wmma_k32(const _Float16* __restrict__ sH,
                                        const _Float16* __restrict__ sW2t,
                                        int rowA, int col, int klo, int koff,
                                        int kbase, v8f acc) {
    v8h a0 = *(const v8h*)&sH[rowA * HID + kbase + klo];
    v8h a1 = *(const v8h*)&sH[rowA * HID + kbase + 16 + klo];
    v16h A = __builtin_shufflevector(a0, a1,
              0,1,2,3,4,5,6,7,8,9,10,11,12,13,14,15);
    v8h b0 = *(const v8h*)&sW2t[col * HID + kbase + (koff >> 1)];
    v8h b1 = *(const v8h*)&sW2t[col * HID + kbase + 16 + (koff >> 1)];
    v16h B = __builtin_shufflevector(b0, b1,
              0,1,2,3,4,5,6,7,8,9,10,11,12,13,14,15);
    return __builtin_amdgcn_wmma_f32_16x16x32_f16(
              false, A, false, B, (short)0, acc, false, false);
}

// Fully fused sphere tracer: one block = 8 rays x 2 channels = 16-row M tile.
// All 25 MLP evals run in-block; layer 2 uses v_wmma_f32_16x16x32_f16.
// Scalar trace state lives in wave-0 registers; all loop-invariant weights
// (layer-1 row, b2[col], layer-3 W3 slice) are register-cached per thread.
__global__ void __launch_bounds__(256)
sphere_trace_kernel(const float* __restrict__ rays_d,
                    const float* __restrict__ rays_o,
                    const float* __restrict__ W1,
                    const float* __restrict__ b1,
                    const _Float16* __restrict__ w2t,
                    const float* __restrict__ b2,
                    const float* __restrict__ W3,
                    const float* __restrict__ b3,
                    float* __restrict__ out)
{
    __shared__ __align__(16) _Float16 sW2t[HID * HID];   // 32 KB, [n][k] f16
    __shared__ __align__(16) _Float16 sH[16 * HID];      // layer-1 act (A matrix)
    __shared__ __align__(16) float    sH2[16 * HID];     // layer-2 act (f32)
    __shared__ __align__(16) float    sPart[256];        // layer-3 partials
    __shared__ __align__(16) float    sPts[16 * 4];      // per-row point xyz

    const int tid = threadIdx.x;

    // ---- stage f16 W2^T into LDS (32 KB, 16-byte coalesced chunks) ----
    {
        const uint4* src = (const uint4*)w2t;
        uint4*       dst = (uint4*)sW2t;
        #pragma unroll
        for (int j = 0; j < 8; ++j) dst[tid + j * 256] = src[tid + j * 256];
    }
    const float b3v = b3[0];

    // ---- per-row (lane<16 of wave 0) register state ----
    float z = 0.f, sdfv = 0.f, nextv = 0.f;
    int   maskv = 1, fixv = 0;
    float dx = 0.f, dy = 0.f, dz = 0.f, ox = 0.f, oy = 0.f, oz = 0.f;
    float sgn = 1.f;
    if (tid < 16) {
        const int rl  = tid & 7;
        const int ray = blockIdx.x * 8 + rl;
        dx = rays_d[ray * 3 + 0]; dy = rays_d[ray * 3 + 1]; dz = rays_d[ray * 3 + 2];
        ox = rays_o[ray * 3 + 0]; oy = rays_o[ray * 3 + 1]; oz = rays_o[ray * 3 + 2];
        sgn = (tid < 8) ? 1.f : -1.f;
    }

    // ---- per-thread WMMA fragment coordinates (layer 2) ----
    const int wave = tid >> 5;
    const int lane = tid & 31;
    const int col  = wave * 16 + (lane & 15);
    const int rowA = lane & 15;
    const int klo  = (lane < 16) ? 0 : 8;    // A frag: K 0..7 / 8..15 (+16..)
    const int koff = (lane < 16) ? 0 : 16;   // B frag: K 0..7 / 8..15 (+16..) via koff>>1
    const int mb   = (lane < 16) ? 0 : 8;    // D frag: rows 0..7 / 8..15

    // ---- register-cache loop-invariant weights (read once from global/L2) ---
    const float cb = b2[col];                            // layer-2 bias
    const int   c8 = (tid & 15) * 8;                     // layer-1 column slice
    float w1x[8], w1y[8], w1z[8], bb1[8];
    {
        const float4 x0 = *(const float4*)&W1[c8 + 0];
        const float4 x1 = *(const float4*)&W1[c8 + 4];
        const float4 y0 = *(const float4*)&W1[HID + c8 + 0];
        const float4 y1 = *(const float4*)&W1[HID + c8 + 4];
        const float4 z0 = *(const float4*)&W1[2 * HID + c8 + 0];
        const float4 z1 = *(const float4*)&W1[2 * HID + c8 + 4];
        const float4 u0 = *(const float4*)&b1[c8 + 0];
        const float4 u1 = *(const float4*)&b1[c8 + 4];
        w1x[0]=x0.x; w1x[1]=x0.y; w1x[2]=x0.z; w1x[3]=x0.w;
        w1x[4]=x1.x; w1x[5]=x1.y; w1x[6]=x1.z; w1x[7]=x1.w;
        w1y[0]=y0.x; w1y[1]=y0.y; w1y[2]=y0.z; w1y[3]=y0.w;
        w1y[4]=y1.x; w1y[5]=y1.y; w1y[6]=y1.z; w1y[7]=y1.w;
        w1z[0]=z0.x; w1z[1]=z0.y; w1z[2]=z0.z; w1z[3]=z0.w;
        w1z[4]=z1.x; w1z[5]=z1.y; w1z[6]=z1.z; w1z[7]=z1.w;
        bb1[0]=u0.x; bb1[1]=u0.y; bb1[2]=u0.z; bb1[3]=u0.w;
        bb1[4]=u1.x; bb1[5]=u1.y; bb1[6]=u1.z; bb1[7]=u1.w;
    }
    const int    q3  = tid & 15;                         // layer-3 W3 slice
    const float4 w3a = *(const float4*)&W3[q3 * 8 + 0];
    const float4 w3b = *(const float4*)&W3[q3 * 8 + 4];

    // 25 evals: ev 0 = initial; then per outer iter: [step-eval, fix0, fix1, fix2]
    for (int ev = 0; ev < 25; ++ev) {
        // ---------- control (wave 0, register state): post(ev-1) + pre(ev) ----
        if (tid < 16) {
            bool upd;
            if (ev == 0) {
                upd = true;                            // z=0 -> pts = origin
            } else {
                // reduce layer-3 partials of eval ev-1
                const float4 p0 = *(const float4*)&sPart[tid * 16 + 0];
                const float4 p1 = *(const float4*)&sPart[tid * 16 + 4];
                const float4 p2 = *(const float4*)&sPart[tid * 16 + 8];
                const float4 p3 = *(const float4*)&sPart[tid * 16 + 12];
                const float evalv = b3v
                    + ((p0.x + p0.y) + (p0.z + p0.w))
                    + ((p1.x + p1.y) + (p1.z + p1.w))
                    + ((p2.x + p2.y) + (p2.z + p2.w))
                    + ((p3.x + p3.y) + (p3.z + p3.w));
                // ---- post-update of eval e = ev-1 ----
                const int e = ev - 1;
                if (e == 0) {
                    nextv = evalv;
                } else {
                    const int psub = (e - 1) & 3;
                    if (psub == 0)      nextv = maskv ? evalv : 0.f;
                    else if (fixv)      nextv = evalv;
                    fixv = (nextv < 0.f);
                    if (psub == 3) {                    // end of outer iter
                        const int rl = tid & 7;
                        const float za = __shfl(z, rl, 32);
                        const float zb = __shfl(z, rl + 8, 32);
                        maskv = maskv && (za < zb);
                    }
                }
                // ---- pre-update of eval ev ----
                const int sub = (ev - 1) & 3;
                if (sub == 0) {                         // sphere-trace step
                    float sv = maskv ? nextv : 0.f;
                    sv = (sv <= THR) ? 0.f : sv;
                    maskv = maskv && (sv > THR);
                    sdfv = sv;
                    z = fmaf(sgn, sv, z);
                    upd = true;
                } else {                                // bisection fix step
                    const float step = 0.5f / (float)(1 << (sub - 1));
                    if (fixv) { z = fmaf(-step * sgn, sdfv, z); upd = true; }
                    else       upd = false;             // points keep old value
                }
            }
            if (upd) {
                sPts[tid * 4 + 0] = fmaf(z, dx, ox);
                sPts[tid * 4 + 1] = fmaf(z, dy, oy);
                sPts[tid * 4 + 2] = fmaf(z, dz, oz);
            }
        }
        __syncthreads();

        // ---------- layer 1: (16x3)@(3x128)+b1, softplus -> f16 A in sH ------
        {
            const int p = tid >> 4;
            const float px = sPts[p * 4 + 0];
            const float py = sPts[p * 4 + 1];
            const float pz = sPts[p * 4 + 2];
            #pragma unroll
            for (int i = 0; i < 8; ++i) {
                const float h = fmaf(px, w1x[i],
                                fmaf(py, w1y[i],
                                fmaf(pz, w1z[i], bb1[i])));
                sH[p * HID + c8 + i] = (_Float16)softplus_f(h);
            }
        }
        __syncthreads();

        // ---------- layer 2: v_wmma_f32_16x16x32_f16, two indep. K-chains ----
        {
            v8f acc0, acc1;
            #pragma unroll
            for (int i = 0; i < 8; ++i) { acc0[i] = cb; acc1[i] = 0.f; }
            acc0 = wmma_k32(sH, sW2t, rowA, col, klo, koff,  0, acc0);
            acc1 = wmma_k32(sH, sW2t, rowA, col, klo, koff, 32, acc1);
            acc0 = wmma_k32(sH, sW2t, rowA, col, klo, koff, 64, acc0);
            acc1 = wmma_k32(sH, sW2t, rowA, col, klo, koff, 96, acc1);
            #pragma unroll
            for (int r = 0; r < 8; ++r)
                sH2[(mb + r) * HID + col] = softplus_f(acc0[r] + acc1[r]);
        }
        __syncthreads();

        // ---------- layer 3: (16x128)@(128x1) partials -> sPart ---------------
        {
            const int p = tid >> 4;
            const float4 h0 = *(const float4*)&sH2[p * HID + q3 * 8 + 0];
            const float4 h1 = *(const float4*)&sH2[p * HID + q3 * 8 + 4];
            float s = h0.x * w3a.x;
            s = fmaf(h0.y, w3a.y, s); s = fmaf(h0.z, w3a.z, s); s = fmaf(h0.w, w3a.w, s);
            s = fmaf(h1.x, w3b.x, s); s = fmaf(h1.y, w3b.y, s);
            s = fmaf(h1.z, w3b.z, s); s = fmaf(h1.w, w3b.w, s);
            sPart[tid] = s;
        }
        __syncthreads();
    }

    // ---------- post(24) + final mask update + outputs (wave 0) ---------------
    if (tid < 16) {
        const float4 p0 = *(const float4*)&sPart[tid * 16 + 0];
        const float4 p1 = *(const float4*)&sPart[tid * 16 + 4];
        const float4 p2 = *(const float4*)&sPart[tid * 16 + 8];
        const float4 p3 = *(const float4*)&sPart[tid * 16 + 12];
        const float evalv = b3v
            + ((p0.x + p0.y) + (p0.z + p0.w))
            + ((p1.x + p1.y) + (p1.z + p1.w))
            + ((p2.x + p2.y) + (p2.z + p2.w))
            + ((p3.x + p3.y) + (p3.z + p3.w));
        // post of eval 24: psub = 3 (last fix of iteration 5)
        if (fixv) nextv = evalv;
        fixv = (nextv < 0.f);
        {
            const int rl = tid & 7;
            const float za = __shfl(z, rl, 32);
            const float zb = __shfl(z, rl + 8, 32);
            maskv = maskv && (za < zb);
        }
        // final top-of-loop mask update
        float sv = maskv ? nextv : 0.f;
        sv = (sv <= THR) ? 0.f : sv;
        maskv = maskv && (sv > THR);

        // outputs: points (2,N,3), z (2,N), mask (2,N) concatenated
        const int ch  = tid >> 3;
        const int rl  = tid & 7;
        const int ray = blockIdx.x * 8 + rl;
        const int PTS = 2 * NRAYS * 3;          // 196608
        float* pout = out + ch * (NRAYS * 3) + ray * 3;
        pout[0] = sPts[tid * 4 + 0];
        pout[1] = sPts[tid * 4 + 1];
        pout[2] = sPts[tid * 4 + 2];
        out[PTS + ch * NRAYS + ray]             = z;
        out[PTS + 2 * NRAYS + ch * NRAYS + ray] = maskv ? 1.0f : 0.0f;
    }
}

extern "C" void kernel_launch(void* const* d_in, const int* in_sizes, int n_in,
                              void* d_out, int out_size, void* d_ws, size_t ws_size,
                              hipStream_t stream) {
    const float* rays_d = (const float*)d_in[0];
    const float* rays_o = (const float*)d_in[1];
    const float* W1     = (const float*)d_in[2];
    const float* b1     = (const float*)d_in[3];
    const float* W2     = (const float*)d_in[4];
    const float* b2     = (const float*)d_in[5];
    const float* W3     = (const float*)d_in[6];
    const float* b3     = (const float*)d_in[7];
    float* out = (float*)d_out;

    _Float16* w2t = (_Float16*)d_ws;            // 128*128 f16 = 32 KB scratch

    prep_w2t_kernel<<<64, 256, 0, stream>>>(W2, w2t);
    sphere_trace_kernel<<<NRAYS / 8, 256, 0, stream>>>(
        rays_d, rays_o, W1, b1, w2t, b2, W3, b3, out);
}
